// Attention_28321014350079
// MI455X (gfx1250) — matrix operand order repaired and checked by reading, not compiled, not run on hardware
//
#include <hip/hip_runtime.h>
#include <math.h>

// ---------------------------------------------------------------------------
// Efficient-attention (PVTv2-style) for MI455X / gfx1250, wave32 + WMMA.
// B=16, C=128, H=W=56 (Nq=3136), heads=4, d=32, Nk=196 (14x14).
// ---------------------------------------------------------------------------

typedef __attribute__((ext_vector_type(16))) _Float16 v16h;
typedef __attribute__((ext_vector_type(8)))  float    v8f;

struct __align__(16) I4 { int x, y, z, w; };

__device__ __forceinline__ int pack2h(float a, float b) {
    union { _Float16 h[2]; int i; } u;
    u.h[0] = (_Float16)a; u.h[1] = (_Float16)b;
    return u.i;
}

// Load 16 packed halves (8 dwords) from LDS as a WMMA operand vector.
__device__ __forceinline__ v16h ld16h(const int* p) {
    union { I4 q[2]; v16h h; } u;
    u.q[0] = ((const I4*)p)[0];
    u.q[1] = ((const I4*)p)[1];
    return u.h;
}

__device__ __forceinline__ v8f wmma_f16(v16h a, v16h b, v8f c) {
    return __builtin_amdgcn_wmma_f32_16x16x32_f16(
        false, a, false, b, (short)0, c, false, false);
}

#define BEPS 1e-5f

// ---------------------------------------------------------------------------
// Kernel 1: depthwise 7x7 stride4 pad3 + BN + GELU(erf) + ch-scale + BN -> h
// grid = B*C (2048), block = 224 (196 active)
// ---------------------------------------------------------------------------
__global__ __launch_bounds__(224) void sr_kernel(
    const float* __restrict__ x,
    const float* __restrict__ sr1_w, const float* __restrict__ sr1_g,
    const float* __restrict__ sr1_b, const float* __restrict__ sr2_w,
    const float* __restrict__ sr2_g, const float* __restrict__ sr2_b,
    float* __restrict__ hbuf) {
    int bc = blockIdx.x;
    int c  = bc & 127;
    int t  = threadIdx.x;
    if (t >= 196) return;
    int oh = t / 14, ow = t % 14;
    const float* xin = x + (long)bc * 3136;
    const float* w   = sr1_w + c * 49;
    float acc = 0.f;
#pragma unroll
    for (int kh = 0; kh < 7; ++kh) {
        int ih = oh * 4 - 3 + kh;
        if (ih < 0 || ih >= 56) continue;
#pragma unroll
        for (int kw = 0; kw < 7; ++kw) {
            int iw = ow * 4 - 3 + kw;
            if (iw < 0 || iw >= 56) continue;
            acc += xin[ih * 56 + iw] * w[kh * 7 + kw];
        }
    }
    float inv = rsqrtf(1.f + BEPS);
    float y = acc * (sr1_g[c] * inv) + sr1_b[c];
    float g = 0.5f * y * (1.f + erff(y * 0.70710678118654752f));
    float z = g * sr2_w[c];
    z = z * (sr2_g[c] * inv) + sr2_b[c];
    hbuf[bc * 196 + t] = z;
}

// ---------------------------------------------------------------------------
// Kernel 2: depthwise 3x3 pad1 + bias + residual -> kv_feat
// ---------------------------------------------------------------------------
__global__ __launch_bounds__(224) void lc_kernel(
    const float* __restrict__ hbuf, const float* __restrict__ lc_w,
    const float* __restrict__ lc_b, float* __restrict__ kvfeat) {
    int bc = blockIdx.x;
    int c  = bc & 127;
    int t  = threadIdx.x;
    if (t >= 196) return;
    int oh = t / 14, ow = t % 14;
    const float* hin = hbuf + bc * 196;
    const float* w   = lc_w + c * 9;
    float acc = lc_b[c];
#pragma unroll
    for (int kh = 0; kh < 3; ++kh) {
        int ih = oh + kh - 1;
        if (ih < 0 || ih >= 14) continue;
#pragma unroll
        for (int kw = 0; kw < 3; ++kw) {
            int iw = ow + kw - 1;
            if (iw < 0 || iw >= 14) continue;
            acc += hin[ih * 14 + iw] * w[kh * 3 + kw];
        }
    }
    kvfeat[bc * 196 + t] = acc + hin[t];
}

// ---------------------------------------------------------------------------
// Kernel 3: kv projection  kv[256,196] = kv_w[256,128] @ kv_feat[128,196]
// per batch, via v_wmma_f32_16x16x32_f16.
// grid = B*2 (32), block = 256 (8 waves, one 16-row o-tile each)
// Writes Kbuf[b,o<128][nk] and Vbuf[b,hd][nk][d].
// ---------------------------------------------------------------------------
__global__ __launch_bounds__(256) void kv_kernel(
    const float* __restrict__ kvfeat, const float* __restrict__ kv_w,
    const float* __restrict__ kv_b, float* __restrict__ Kbuf,
    float* __restrict__ Vbuf) {
    __shared__ int sB[52 * 256];   // kv_feat in WMMA-B layout: 4 kc x 13 nt
    int b     = blockIdx.x >> 1;
    int mhalf = blockIdx.x & 1;
    int tid   = threadIdx.x;
    const float* feat = kvfeat + b * 128 * 196;

    for (int idx = tid; idx < 52 * 256; idx += 256) {
        int unit = idx >> 8;            // 0..51
        int kc = unit / 13, nt = unit % 13;
        int r = idx & 255;
        int lane = r >> 3, j = r & 7;
        int n = lane & 15, khalf = (lane >> 4) << 4;
        int cc = kc * 32 + (khalf >> 1) + 2 * j + ((j & 4) << 1);
        int nk = nt * 16 + n;
        float f0 = 0.f, f1 = 0.f;
        if (nk < 196) { f0 = feat[cc * 196 + nk]; f1 = feat[(cc + 1) * 196 + nk]; }
        sB[idx] = pack2h(f0, f1);
    }
    __syncthreads();

    int w = tid >> 5, lane = tid & 31;
    int obase = (mhalf * 8 + w) * 16;
    int m = lane & 15;
    int kb = (lane >> 4) << 3;
    int n = m;

    v8f zero = {};
    v8f acc[13];
#pragma unroll
    for (int t = 0; t < 13; ++t) acc[t] = zero;

#pragma unroll
    for (int kc = 0; kc < 4; ++kc) {
        union { _Float16 h[16]; v16h v; } ua;   // kv_w tile as A operand
#pragma unroll
        for (int hi = 0; hi < 16; ++hi) {
            int k = kb + (hi < 8 ? hi : hi + 8);  // A layout: K 0..7 then 16..23
            ua.h[hi] = (_Float16)kv_w[(obase + m) * 128 + kc * 32 + k];
        }
#pragma unroll
        for (int nt = 0; nt < 13; ++nt) {
            v16h bm = ld16h(&sB[(kc * 13 + nt) * 256 + lane * 8]);
            acc[nt] = wmma_f16(ua.v, bm, acc[nt]);
        }
    }

#pragma unroll
    for (int r = 0; r < 8; ++r) {
        int o = obase + r + ((lane >> 4) << 3);
        float bias = kv_b[o];
#pragma unroll
        for (int nt = 0; nt < 13; ++nt) {
            int nk = nt * 16 + n;
            if (nk < 196) {
                float val = acc[nt][r] + bias;
                if (o < 128) {
                    Kbuf[(b * 128 + o) * 196 + nk] = val;
                } else {
                    int d = o - 128;
                    Vbuf[((b * 4 + (d >> 5)) * 196 + nk) * 32 + (d & 31)] = val;
                }
            }
        }
    }
}

// ---------------------------------------------------------------------------
// Kernel 4: q-projection + S=QK^T*scale + rpe + softmax + P@V
// grid = B*4*25 (1600), block = 256 (8 waves; wave = one 16-row q tile)
// ---------------------------------------------------------------------------
__global__ __launch_bounds__(256) void attn_kernel(
    const float* __restrict__ x, const float* __restrict__ rpe,
    const float* __restrict__ q_w, const float* __restrict__ q_b,
    const float* __restrict__ Kbuf, const float* __restrict__ Vbuf,
    float* __restrict__ out) {
    __shared__ int sKb[13 * 256];             // K  (B layout), 13 nk-tiles
    __shared__ int sVb[14 * 256];             // V  (B layout), 7 kc x 2 d-tiles
    __shared__ int sQw[8 * 256];              // q_w (B layout), 4 kc x 2 d-tiles
    __shared__ __align__(16) _Float16 sScr[8][16 * 32];  // per-wave relayout

    int qg  = blockIdx.x % 25;
    int hd  = (blockIdx.x / 25) & 3;
    int b   = blockIdx.x / 100;
    int tid = threadIdx.x;

    const float* Khead = Kbuf + (b * 128 + hd * 32) * 196;   // [d][nk]
    const float* Vhead = Vbuf + (b * 4 + hd) * 196 * 32;     // [nk][d]

    for (int idx = tid; idx < 13 * 256; idx += 256) {
        int t = idx >> 8, r = idx & 255;
        int lane = r >> 3, j = r & 7;
        int n = lane & 15, khalf = (lane >> 4) << 4;
        int k0 = (khalf >> 1) + 2 * j + ((j & 4) << 1);
        int nk = t * 16 + n;
        float f0 = 0.f, f1 = 0.f;
        if (nk < 196) { f0 = Khead[k0 * 196 + nk]; f1 = Khead[(k0 + 1) * 196 + nk]; }
        sKb[idx] = pack2h(f0, f1);
    }
    for (int idx = tid; idx < 14 * 256; idx += 256) {
        int u = idx >> 8;                    // kc*2 + d-tile
        int kc = u >> 1, nt = u & 1;
        int r = idx & 255;
        int lane = r >> 3, j = r & 7;
        int n = lane & 15, khalf = (lane >> 4) << 4;
        int nk = kc * 32 + (khalf >> 1) + 2 * j + ((j & 4) << 1);
        int d  = nt * 16 + n;
        float f0 = (nk     < 196) ? Vhead[nk * 32 + d]       : 0.f;
        float f1 = (nk + 1 < 196) ? Vhead[(nk + 1) * 32 + d] : 0.f;
        sVb[idx] = pack2h(f0, f1);
    }
    for (int idx = tid; idx < 8 * 256; idx += 256) {
        int u = idx >> 8;
        int kc = u >> 1, nt = u & 1;
        int r = idx & 255;
        int lane = r >> 3, j = r & 7;
        int n = lane & 15, khalf = (lane >> 4) << 4;
        int o  = hd * 32 + nt * 16 + n;
        int c0 = kc * 32 + (khalf >> 1) + 2 * j + ((j & 4) << 1);
        sQw[idx] = pack2h(q_w[o * 128 + c0], q_w[o * 128 + c0 + 1]);
    }
    __syncthreads();

    int w = tid >> 5, lane = tid & 31;
    int qtile = qg * 8 + w;
    if (qtile >= 196) return;                // tail waves idle (no barriers below)
    int qbase = qtile * 16;
    int m  = lane & 15;
    int kb = (lane >> 4) << 3;
    int n  = m;
    _Float16* scr = &sScr[w][0];

    const float* rpr = rpe + (hd * 3136 + qbase) * 196;
    __builtin_prefetch(rpr + m * 196, 0, 3);   // warm rpe rows (L2 resident)

    // ---- q projection: Q(16x32) = x-tile(16x128) @ q_w^T, scale+bias folded
    const float* xb = x + (long)b * 128 * 3136;
    int pix = qbase + m;
    v8f zero = {};
    v8f accQ0 = zero, accQ1 = zero;
#pragma unroll
    for (int kc = 0; kc < 4; ++kc) {
        union { _Float16 h[16]; v16h v; } ua;
#pragma unroll
        for (int hi = 0; hi < 16; ++hi) {
            int k = kb + (hi < 8 ? hi : hi + 8);
            ua.h[hi] = (_Float16)xb[(kc * 32 + k) * 3136 + pix];
        }
        v16h b0 = ld16h(&sQw[(kc * 2 + 0) * 256 + lane * 8]);
        accQ0 = wmma_f16(ua.v, b0, accQ0);
        v16h b1 = ld16h(&sQw[(kc * 2 + 1) * 256 + lane * 8]);
        accQ1 = wmma_f16(ua.v, b1, accQ1);
    }
    const float scale = 0.17677669529663687f;  // 1/sqrt(32)
    float bias0 = q_b[hd * 32 + n];
    float bias1 = q_b[hd * 32 + 16 + n];
#pragma unroll
    for (int r = 0; r < 8; ++r) {              // C layout -> scratch (row-major)
        int mm = r + ((lane >> 4) << 3);
        scr[mm * 32 + n]      = (_Float16)((accQ0[r] + bias0) * scale);
        scr[mm * 32 + 16 + n] = (_Float16)((accQ1[r] + bias1) * scale);
    }
    asm volatile("s_wait_dscnt 0" ::: "memory");
    v16h qa;
    {
        union { I4 q[2]; v16h h; } u;          // A layout: K 0..7 | 16..23
        u.q[0] = *(const I4*)(scr + m * 32 + kb);
        u.q[1] = *(const I4*)(scr + m * 32 + kb + 16);
        qa = u.h;
    }

    // ---- S = Q @ K  (13 nk-tiles, K-dim 32 in a single WMMA each)
    v8f s[13];
#pragma unroll
    for (int t = 0; t < 13; ++t) {
        v16h kbm = ld16h(&sKb[t * 256 + lane * 8]);
        s[t] = wmma_f16(qa, kbm, zero);
    }

    // ---- + rpe, mask, softmax over nk (row = 16 lanes of one half-wave)
#pragma unroll
    for (int t = 0; t < 13; ++t) {
        int nk = t * 16 + n;
        bool valid = (nk < 196);
#pragma unroll
        for (int r = 0; r < 8; ++r) {
            int mm = r + ((lane >> 4) << 3);
            float v = s[t][r];
            v = valid ? (v + rpr[mm * 196 + nk]) : -1e30f;
            s[t][r] = v;
        }
    }
    float mx[8], sm[8];
#pragma unroll
    for (int r = 0; r < 8; ++r) {
        float v = s[0][r];
#pragma unroll
        for (int t = 1; t < 13; ++t) v = fmaxf(v, s[t][r]);
        v = fmaxf(v, __shfl_xor(v, 1, 32));
        v = fmaxf(v, __shfl_xor(v, 2, 32));
        v = fmaxf(v, __shfl_xor(v, 4, 32));
        v = fmaxf(v, __shfl_xor(v, 8, 32));
        mx[r] = v;
    }
#pragma unroll
    for (int r = 0; r < 8; ++r) {
        float acc = 0.f;
#pragma unroll
        for (int t = 0; t < 13; ++t) {
            float p = __expf(s[t][r] - mx[r]);
            s[t][r] = p;
            acc += p;
        }
        acc += __shfl_xor(acc, 1, 32);
        acc += __shfl_xor(acc, 2, 32);
        acc += __shfl_xor(acc, 4, 32);
        acc += __shfl_xor(acc, 8, 32);
        sm[r] = acc;
    }
#pragma unroll
    for (int r = 0; r < 8; ++r) {
        float inv = 1.0f / sm[r];
#pragma unroll
        for (int t = 0; t < 13; ++t) s[t][r] = s[t][r] * inv;
    }

    // ---- O = P @ V : 7 chunks of K-dim 32 (P relayout C->A via scratch)
    v8f o0 = zero, o1 = zero;
#pragma unroll
    for (int kc = 0; kc < 7; ++kc) {
        int t0 = 2 * kc, t1 = 2 * kc + 1;
#pragma unroll
        for (int r = 0; r < 8; ++r) {
            int mm = r + ((lane >> 4) << 3);
            scr[mm * 32 + n]      = (_Float16)s[t0][r];
            scr[mm * 32 + 16 + n] = (t1 < 13) ? (_Float16)s[t1][r] : (_Float16)0.f;
        }
        asm volatile("s_wait_dscnt 0" ::: "memory");
        union { I4 q[2]; v16h h; } u;
        u.q[0] = *(const I4*)(scr + m * 32 + kb);
        u.q[1] = *(const I4*)(scr + m * 32 + kb + 16);
        v16h vb0 = ld16h(&sVb[(kc * 2 + 0) * 256 + lane * 8]);
        o0 = wmma_f16(u.h, vb0, o0);
        v16h vb1 = ld16h(&sVb[(kc * 2 + 1) * 256 + lane * 8]);
        o1 = wmma_f16(u.h, vb1, o1);
    }

    // ---- store out[b][hd*32 + d][qbase + m]
    float* ob = out + ((long)b * 128 + hd * 32) * 3136 + qbase;
#pragma unroll
    for (int r = 0; r < 8; ++r) {
        int mm = r + ((lane >> 4) << 3);
        ob[n * 3136 + mm]        = o0[r];
        ob[(16 + n) * 3136 + mm] = o1[r];
    }
}

// ---------------------------------------------------------------------------
extern "C" void kernel_launch(void* const* d_in, const int* in_sizes, int n_in,
                              void* d_out, int out_size, void* d_ws, size_t ws_size,
                              hipStream_t stream) {
    (void)in_sizes; (void)n_in; (void)out_size; (void)ws_size;
    const float* x     = (const float*)d_in[0];
    const float* rpe   = (const float*)d_in[1];
    const float* q_w   = (const float*)d_in[2];
    const float* q_b   = (const float*)d_in[3];
    const float* kv_w  = (const float*)d_in[4];
    const float* kv_b  = (const float*)d_in[5];
    const float* sr1_w = (const float*)d_in[6];
    const float* sr1_g = (const float*)d_in[7];
    const float* sr1_b = (const float*)d_in[8];
    const float* sr2_w = (const float*)d_in[9];
    const float* sr2_g = (const float*)d_in[10];
    const float* sr2_b = (const float*)d_in[11];
    const float* lc_w  = (const float*)d_in[12];
    const float* lc_b  = (const float*)d_in[13];
    float* out = (float*)d_out;

    float* ws     = (float*)d_ws;
    float* hbuf   = ws;                 // 16*128*196 = 401408 floats
    float* kvfeat = ws + 401408;        // 401408
    float* Kbuf   = ws + 802816;        // 401408  [b][o<128][nk]
    float* Vbuf   = ws + 1204224;       // 401408  [b][hd][nk][d]

    sr_kernel<<<dim3(2048), dim3(224), 0, stream>>>(
        x, sr1_w, sr1_g, sr1_b, sr2_w, sr2_g, sr2_b, hbuf);
    lc_kernel<<<dim3(2048), dim3(224), 0, stream>>>(hbuf, lc_w, lc_b, kvfeat);
    kv_kernel<<<dim3(32), dim3(256), 0, stream>>>(kvfeat, kv_w, kv_b, Kbuf, Vbuf);
    attn_kernel<<<dim3(1600), dim3(256), 0, stream>>>(
        x, rpe, q_w, q_b, Kbuf, Vbuf, out);
}
